// SSRPAttributor_68204080661076
// MI455X (gfx1250) — matrix operand order repaired and checked by reading, not compiled, not run on hardware
//
#include <hip/hip_runtime.h>
#include <hip/hip_bf16.h>

typedef __attribute__((ext_vector_type(16))) _Float16 v16h;
typedef __attribute__((ext_vector_type(8)))  _Float16 v8h;
typedef __attribute__((ext_vector_type(4)))  _Float16 v4h;
typedef __attribute__((ext_vector_type(8)))  float    v8f;
typedef __attribute__((ext_vector_type(4)))  float    v4f;

#define BATCH 8
#define HEADS 12
#define SEQ   2048
#define DIM   64
#define RR    64
#define MM    256
#define EPSC  1e-6f
#define LAMC  0.5f

// -------- workspace layout (float offsets) --------
#define OFF_KV     ((size_t)0)                        // 96*4096
#define OFF_U      (OFF_KV + 96*4096)                 // 96*4096
#define OFF_WT     (OFF_U + 96*4096)                  // 96*4096
#define OFF_S      (OFF_WT + 96*4096)                 // 96*64
#define OFF_SSUM   (OFF_S + 96*64)                    // 96
#define OFF_GPART  (OFF_SSUM + 96)                    // 96*64  (zeroed)
#define OFF_COLSUM (OFF_GPART + 96*64)                // 96*64  (zeroed, contiguous w/ GPART)
#define OFF_CC     (OFF_COLSUM + 96*64)               // 96*64
#define OFF_A0     (OFF_CC + 96*64)                   // 4096
#define OFF_LCH    (OFF_A0 + 4096)                    // 4096
#define OFF_RAW    (OFF_LCH + 4096)                   // BATCH*SEQ (zeroed)

__device__ __forceinline__ float elu1(float x) {
  return x > 0.f ? x + 1.f : __expf(x);
}
__device__ __forceinline__ v16h cat16(v8h lo, v8h hi) {
  return __builtin_shufflevector(lo, hi, 0, 1, 2, 3, 4, 5, 6, 7,
                                 8, 9, 10, 11, 12, 13, 14, 15);
}
__device__ __forceinline__ v8f wmma16(v16h a, v16h b, v8f c) {
  return __builtin_amdgcn_wmma_f32_16x16x32_f16(false, a, false, b, (short)0, c,
                                                false, false);
}
// Fragment geometry (wave32, 16x16x32 f16 WMMA):
//  A (16x32): lane m=lane&15; halves are two contiguous runs of 8 at
//             k = (lane>>4)*8 and 16+(lane>>4)*8.
//  B (32x16): lane col=lane&15; halves are two contiguous runs of 8 at
//             k = (lane>>4)*8 and 16+(lane>>4)*8 (same as A).
//  C/D f32:   element i -> row m = (lane>>4)*8 + i, col = lane&15.

// ============ K1: kv[b,h] = (elu(k)+1)^T @ v  (64x64x2048) ============
__global__ __launch_bounds__(256) void kv_kernel(const float* __restrict__ kin,
                                                 const float* __restrict__ vin,
                                                 float* __restrict__ kv) {
  const int bh = blockIdx.x;
  const float* kp = kin + (size_t)bh * SEQ * DIM;
  const float* vp = vin + (size_t)bh * SEQ * DIM;
  // transposed tiles: [d or e][n-in-chunk], row stride 40 halves = 80B (16B mult)
  __shared__ alignas(16) _Float16 ldsKT[64][40];
  __shared__ alignas(16) _Float16 ldsVT[64][40];
  const int tid = threadIdx.x, lane = tid & 31, wave = tid >> 5;
  const int dt = (wave >> 1) * 16;      // row tile (d)
  const int et0 = (wave & 1) * 32;      // first col tile (e)
  const int m = lane & 15;
  const int klo = (lane >> 4) * 8;      // A-frag k base
  const int kb = (lane >> 4) * 16;      // unused; B frags also use klo
  v8f acc0 = {}, acc1 = {};
  for (int n0 = 0; n0 < SEQ; n0 += 32) {
#pragma unroll
    for (int i = 0; i < 2; ++i) {
      int idx = (tid + 256 * i) * 4;    // 2048 elems as float4
      int r = idx >> 6, c = idx & 63;
      v4f kvv = *(const v4f*)&kp[(n0 + r) * DIM + c];
      v4f vvv = *(const v4f*)&vp[(n0 + r) * DIM + c];
#pragma unroll
      for (int e = 0; e < 4; ++e) {
        ldsKT[c + e][r] = (_Float16)elu1(kvv[e]);
        ldsVT[c + e][r] = (_Float16)vvv[e];
      }
    }
    if (n0 + 32 < SEQ) {
      __builtin_prefetch(&kp[(n0 + 32) * DIM + tid * 8], 0, 0);
      __builtin_prefetch(&vp[(n0 + 32) * DIM + tid * 8], 0, 0);
    }
    __syncthreads();
    v16h a = cat16(*(const v8h*)&ldsKT[dt + m][klo],
                   *(const v8h*)&ldsKT[dt + m][16 + klo]);
    v16h b0 = cat16(*(const v8h*)&ldsVT[et0 + m][klo],
                    *(const v8h*)&ldsVT[et0 + m][16 + klo]);
    v16h b1 = cat16(*(const v8h*)&ldsVT[et0 + 16 + m][klo],
                    *(const v8h*)&ldsVT[et0 + 16 + m][16 + klo]);
    acc0 = wmma16(a, b0, acc0);
    acc1 = wmma16(a, b1, acc1);
    __syncthreads();
  }
  float* kvp = kv + (size_t)bh * 64 * 64;
#pragma unroll
  for (int i = 0; i < 8; ++i) {
    int mrow = ((lane >> 4) << 3) + i;
    kvp[(dt + mrow) * 64 + et0 + m]      = acc0[i];
    kvp[(dt + mrow) * 64 + et0 + 16 + m] = acc1[i];
  }
}

// ============ K2: one-sided Jacobi SVD of kv (64x64) per head ============
__global__ __launch_bounds__(256) void svd_kernel(const float* __restrict__ kv,
                                                  float* __restrict__ Uo,
                                                  float* __restrict__ Wt,
                                                  float* __restrict__ So,
                                                  float* __restrict__ Ssum) {
  const int bh = blockIdx.x;
  __shared__ float G[64][65];
  __shared__ float V[64][65];
  __shared__ float Scol[64];
  const int tid = threadIdx.x;
#pragma unroll
  for (int i = 0; i < 16; ++i) {
    int idx = tid + 256 * i;
    int r = idx >> 6, c = idx & 63;
    G[r][c] = kv[(size_t)bh * 4096 + idx];
    V[r][c] = (r == c) ? 1.f : 0.f;
  }
  __syncthreads();
  const int grp = tid >> 3, sub = tid & 7;
  for (int sweep = 0; sweep < 10; ++sweep) {
    for (int t = 0; t < 63; ++t) {
      int p, q;
      if (grp == 0) { p = 0; q = (t % 63) + 1; }
      else { p = ((grp + t) % 63) + 1; q = ((63 - grp + t) % 63) + 1; }
      if (p > q) { int tmp = p; p = q; q = tmp; }
      float app = 0.f, aqq = 0.f, apq = 0.f;
      for (int i = sub; i < 64; i += 8) {
        float gp = G[i][p], gq = G[i][q];
        app += gp * gp; aqq += gq * gq; apq += gp * gq;
      }
#pragma unroll
      for (int msk = 1; msk < 8; msk <<= 1) {
        app += __shfl_xor(app, msk, 32);
        aqq += __shfl_xor(aqq, msk, 32);
        apq += __shfl_xor(apq, msk, 32);
      }
      float c = 1.f, s = 0.f;
      if (fabsf(apq) > 1e-12f * sqrtf(app * aqq) + 1e-30f) {
        float tau = (aqq - app) / (2.f * apq);
        float tt = (tau >= 0.f ? 1.f : -1.f) / (fabsf(tau) + sqrtf(1.f + tau * tau));
        c = rsqrtf(1.f + tt * tt);
        s = tt * c;
      }
      for (int i = sub; i < 64; i += 8) {
        float gp = G[i][p], gq = G[i][q];
        G[i][p] = c * gp - s * gq; G[i][q] = s * gp + c * gq;
        float xp = V[i][p], xq = V[i][q];
        V[i][p] = c * xp - s * xq; V[i][q] = s * xp + c * xq;
      }
      __syncthreads();
    }
  }
  const int col = tid >> 2, s4 = tid & 3;
  float ss = 0.f;
  for (int i = s4; i < 64; i += 4) ss += G[i][col] * G[i][col];
#pragma unroll
  for (int msk = 1; msk < 4; msk <<= 1) ss += __shfl_xor(ss, msk, 32);
  if (s4 == 0) Scol[col] = sqrtf(ss);
  __syncthreads();
#pragma unroll
  for (int i = 0; i < 16; ++i) {
    int idx = tid + 256 * i;
    int d = idx >> 6, r = idx & 63;
    Uo[(size_t)bh * 4096 + idx] = G[d][r] / fmaxf(Scol[r], 1e-20f);
    Wt[(size_t)bh * 4096 + idx] = V[r][d];   // Wt[d][r] = V[r][d] so proj = wb @ Wt
  }
  if (tid < 64) So[bh * 64 + tid] = Scol[tid];
  if (tid < 32) {
    float x = Scol[tid] + Scol[tid + 32];
#pragma unroll
    for (int msk = 1; msk < 32; msk <<= 1) x += __shfl_xor(x, msk, 32);
    if (tid == 0) Ssum[bh] = x;
  }
}

// ============ K3: z = qf@U, proj = wb@Wt; accumulate g & colsum ============
__global__ __launch_bounds__(256) void zproj_kernel(const float* __restrict__ qin,
                                                    const float* __restrict__ wbin,
                                                    const float* __restrict__ Uo,
                                                    const float* __restrict__ Wt,
                                                    float* __restrict__ gpart,
                                                    float* __restrict__ colsum) {
  const int blk = blockIdx.x;
  const int bh = blk >> 4;
  const int n0 = (blk & 15) * 128;
  const float* qp = qin + (size_t)bh * SEQ * DIM;
  const float* wp = wbin + (size_t)bh * SEQ * DIM;
  __shared__ alignas(16) _Float16 ldsQ[128][72];   // row-major (A side)
  __shared__ alignas(16) _Float16 ldsB[128][72];
  __shared__ alignas(16) _Float16 ldsUT[64][72];   // transposed: [r][d] (B side)
  __shared__ alignas(16) _Float16 ldsWT[64][72];
  __shared__ float sg[64], sc[64];
  const int tid = threadIdx.x, lane = tid & 31, wave = tid >> 5;
  const int m = lane & 15;
  const int klo = (lane >> 4) * 8;
  const int kb = (lane >> 4) * 16;
  if (tid < 64) { sg[tid] = 0.f; sc[tid] = 0.f; }
#pragma unroll
  for (int i = 0; i < 8; ++i) {
    int idx = (tid + 256 * i) * 4;      // 8192 elems as float4
    int r = idx >> 6, c = idx & 63;
    v4f qv = *(const v4f*)&qp[(n0 + r) * DIM + c];
    v4f wv = *(const v4f*)&wp[(n0 + r) * DIM + c];
    v4h qh, wh;
#pragma unroll
    for (int e = 0; e < 4; ++e) {
      qh[e] = (_Float16)elu1(qv[e]);
      wh[e] = (_Float16)wv[e];
    }
    *(v4h*)&ldsQ[r][c] = qh;
    *(v4h*)&ldsB[r][c] = wh;
  }
#pragma unroll
  for (int i = 0; i < 4; ++i) {
    int idx = (tid + 256 * i) * 4;      // 4096 elems as float4
    int d = idx >> 6, r = idx & 63;
    v4f uv = *(const v4f*)&Uo[(size_t)bh * 4096 + idx];
    v4f wv = *(const v4f*)&Wt[(size_t)bh * 4096 + idx];
#pragma unroll
    for (int e = 0; e < 4; ++e) {
      ldsUT[r + e][d] = (_Float16)uv[e];
      ldsWT[r + e][d] = (_Float16)wv[e];
    }
  }
  __syncthreads();
  v8f az[4] = {}, ap[4] = {};
  const int arow = wave * 16 + m;
#pragma unroll
  for (int ch = 0; ch < 2; ++ch) {
    v16h aQ = cat16(*(const v8h*)&ldsQ[arow][ch * 32 + klo],
                    *(const v8h*)&ldsQ[arow][ch * 32 + 16 + klo]);
    v16h aW = cat16(*(const v8h*)&ldsB[arow][ch * 32 + klo],
                    *(const v8h*)&ldsB[arow][ch * 32 + 16 + klo]);
#pragma unroll
    for (int ct = 0; ct < 4; ++ct) {
      v16h bU = cat16(*(const v8h*)&ldsUT[ct * 16 + m][ch * 32 + klo],
                      *(const v8h*)&ldsUT[ct * 16 + m][ch * 32 + 16 + klo]);
      v16h bW = cat16(*(const v8h*)&ldsWT[ct * 16 + m][ch * 32 + klo],
                      *(const v8h*)&ldsWT[ct * 16 + m][ch * 32 + 16 + klo]);
      az[ct] = wmma16(aQ, bU, az[ct]);
      ap[ct] = wmma16(aW, bW, ap[ct]);
    }
  }
#pragma unroll
  for (int ct = 0; ct < 4; ++ct) {
    float ps = 0.f, zs = 0.f;
#pragma unroll
    for (int i = 0; i < 8; ++i) {
      float z = az[ct][i], p = ap[ct][i];
      ps += z * p;
      zs += fmaxf(z, 0.f);
    }
    ps += __shfl_xor(ps, 16, 32);
    zs += __shfl_xor(zs, 16, 32);
    if (lane < 16) {
      atomicAdd(&sg[ct * 16 + lane], ps);
      atomicAdd(&sc[ct * 16 + lane], zs);
    }
  }
  __syncthreads();
  if (tid < 64) {
    atomicAdd(&gpart[bh * 64 + tid], sg[tid]);
    atomicAdd(&colsum[bh * 64 + tid], sc[tid]);
  }
}

// ============ K4a: A0 = masks^T diag(wgt) masks ============
__global__ __launch_bounds__(256) void amat_kernel(const float* __restrict__ masks,
                                                   float* __restrict__ A0,
                                                   float* __restrict__ Ach) {
  __shared__ float wg[MM];
  const int tid = threadIdx.x;
  float s = 0.f;
  for (int r = 0; r < RR; ++r) s += masks[tid * RR + r];
  wg[tid] = (RR - 1.f) / (s * (RR - s));
  __syncthreads();
  for (int e = 0; e < 16; ++e) {
    int idx = tid + 256 * e;
    int i = idx >> 6, j = idx & 63;
    float a = 0.f;
    for (int mi = 0; mi < MM; ++mi)
      a += masks[mi * RR + i] * wg[mi] * masks[mi * RR + j];
    A0[idx] = a;
    Ach[idx] = a + (i == j ? EPSC : 0.f);
  }
}

// ============ K4b: in-place Cholesky of A (64x64) ============
__global__ __launch_bounds__(64) void chol_kernel(float* __restrict__ Ach) {
  __shared__ float Am[64][65];
  const int tid = threadIdx.x;
  for (int i = 0; i < 64; ++i) Am[i][tid] = Ach[i * 64 + tid];
  __syncthreads();
  for (int j = 0; j < 64; ++j) {
    if (tid == j) Am[j][j] = sqrtf(fmaxf(Am[j][j], 1e-20f));
    __syncthreads();
    if (tid > j) Am[tid][j] /= Am[j][j];
    __syncthreads();
    if (tid > j) {
      float lij = Am[tid][j];
      for (int kk = j + 1; kk <= tid; ++kk) Am[tid][kk] -= lij * Am[kk][j];
    }
    __syncthreads();
  }
  for (int i = 0; i < 64; ++i) Ach[i * 64 + tid] = Am[i][tid];
}

// ============ K4c: per-head R_spec, Shapley solve, combined coef ============
__global__ __launch_bounds__(64) void spec_kernel(const float* __restrict__ gpart,
                                                  const float* __restrict__ So,
                                                  const float* __restrict__ Ft,
                                                  const float* __restrict__ colsum,
                                                  const float* __restrict__ Ssum,
                                                  const float* __restrict__ A0,
                                                  const float* __restrict__ Lch,
                                                  float* __restrict__ cc) {
  const int bh = blockIdx.x, b = bh / HEADS, tid = threadIdx.x;
  __shared__ float gl[64], rhs[64], phi[64], yS[64], red[64], Ll[64][65];
  for (int i = 0; i < 64; ++i) Ll[i][tid] = Lch[i * 64 + tid];
  float gv = So[bh * 64 + tid] * gpart[bh * 64 + tid];
  gl[tid] = gv;
  red[tid] = fabsf(gv) + EPSC;
  __syncthreads();
  for (int off = 32; off >= 1; off >>= 1) {
    if (tid < off) red[tid] += red[tid + off];
    __syncthreads();
  }
  const float ftb = Ft[b];
  const float rspec = (fabsf(gv) + EPSC) / red[0] * ftb;
  float a = 0.f;
  for (int j = 0; j < 64; ++j) a += A0[tid * 64 + j] * gl[j];
  rhs[tid] = a;
  __syncthreads();
  if (tid == 0) {
    for (int i = 0; i < 64; ++i) {
      float acc = rhs[i];
      for (int kk = 0; kk < i; ++kk) acc -= Ll[i][kk] * yS[kk];
      yS[i] = acc / Ll[i][i];
    }
    for (int i = 63; i >= 0; --i) {
      float acc = yS[i];
      for (int kk = i + 1; kk < 64; ++kk) acc -= Ll[kk][i] * phi[kk];
      phi[i] = acc / Ll[i][i];
    }
  }
  __syncthreads();
  float ssb = 0.f;
  for (int hh = 0; hh < HEADS; ++hh) ssb += Ssum[b * HEADS + hh];
  const float hw = Ssum[bh] / (ssb + EPSC);
  const float coef = hw * ((1.f - LAMC) * rspec + LAMC * phi[tid]);
  cc[bh * 64 + tid] = coef / (colsum[bh * 64 + tid] + EPSC);
}

// ============ K5: raw[b,n] += sum_r max(z,0)*cc[r]  (recompute z via WMMA) ============
__global__ __launch_bounds__(256) void out_kernel(const float* __restrict__ qin,
                                                  const float* __restrict__ Uo,
                                                  const float* __restrict__ cc,
                                                  float* __restrict__ raw) {
  const int blk = blockIdx.x;
  const int bh = blk >> 4;
  const int b = bh / HEADS;
  const int n0 = (blk & 15) * 128;
  const float* qp = qin + (size_t)bh * SEQ * DIM;
  __shared__ alignas(16) _Float16 ldsQ[128][72];
  __shared__ alignas(16) _Float16 ldsUT[64][72];
  __shared__ float ccL[64];
  const int tid = threadIdx.x, lane = tid & 31, wave = tid >> 5;
  const int m = lane & 15;
  const int klo = (lane >> 4) * 8;
  const int kb = (lane >> 4) * 16;
#pragma unroll
  for (int i = 0; i < 8; ++i) {
    int idx = (tid + 256 * i) * 4;
    int r = idx >> 6, c = idx & 63;
    v4f qv = *(const v4f*)&qp[(n0 + r) * DIM + c];
    v4h qh;
#pragma unroll
    for (int e = 0; e < 4; ++e) qh[e] = (_Float16)elu1(qv[e]);
    *(v4h*)&ldsQ[r][c] = qh;
  }
#pragma unroll
  for (int i = 0; i < 4; ++i) {
    int idx = (tid + 256 * i) * 4;
    int d = idx >> 6, r = idx & 63;
    v4f uv = *(const v4f*)&Uo[(size_t)bh * 4096 + idx];
#pragma unroll
    for (int e = 0; e < 4; ++e) ldsUT[r + e][d] = (_Float16)uv[e];
  }
  if (tid < 64) ccL[tid] = cc[bh * 64 + tid];
  __syncthreads();
  v8f az[4] = {};
  const int arow = wave * 16 + m;
#pragma unroll
  for (int ch = 0; ch < 2; ++ch) {
    v16h aQ = cat16(*(const v8h*)&ldsQ[arow][ch * 32 + klo],
                    *(const v8h*)&ldsQ[arow][ch * 32 + 16 + klo]);
#pragma unroll
    for (int ct = 0; ct < 4; ++ct) {
      v16h bU = cat16(*(const v8h*)&ldsUT[ct * 16 + m][ch * 32 + klo],
                      *(const v8h*)&ldsUT[ct * 16 + m][ch * 32 + 16 + klo]);
      az[ct] = wmma16(aQ, bU, az[ct]);
    }
  }
#pragma unroll
  for (int i = 0; i < 8; ++i) {
    float acc = 0.f;
#pragma unroll
    for (int ct = 0; ct < 4; ++ct)
      acc += fmaxf(az[ct][i], 0.f) * ccL[ct * 16 + m];
#pragma unroll
    for (int msk = 1; msk < 16; msk <<= 1) acc += __shfl_xor(acc, msk, 32);
    if (m == 0) {
      int n = n0 + wave * 16 + ((lane >> 4) << 3) + i;
      atomicAdd(&raw[b * SEQ + n], acc);
    }
  }
}

// ============ K6: double clamp + renormalize per batch ============
__global__ __launch_bounds__(256) void norm_kernel(const float* __restrict__ raw,
                                                   const float* __restrict__ Ft,
                                                   float* __restrict__ out) {
  const int b = blockIdx.x, tid = threadIdx.x;
  __shared__ float red[256];
  float vals[8];
  float s1 = 0.f;
#pragma unroll
  for (int i = 0; i < 8; ++i) {
    float x = fmaxf(raw[b * SEQ + tid + 256 * i], 0.f);
    vals[i] = x; s1 += x;
  }
  red[tid] = s1; __syncthreads();
  for (int off = 128; off >= 1; off >>= 1) {
    if (tid < off) red[tid] += red[tid + off];
    __syncthreads();
  }
  const float ftb = Ft[b];
  const float sc1 = ftb / (red[0] + EPSC);
  __syncthreads();
  float s2 = 0.f;
#pragma unroll
  for (int i = 0; i < 8; ++i) {
    float x = fmaxf(vals[i] * sc1, 0.f);
    vals[i] = x; s2 += x;
  }
  red[tid] = s2; __syncthreads();
  for (int off = 128; off >= 1; off >>= 1) {
    if (tid < off) red[tid] += red[tid + off];
    __syncthreads();
  }
  const float sc2 = ftb / (red[0] + EPSC);
#pragma unroll
  for (int i = 0; i < 8; ++i)
    out[b * SEQ + tid + 256 * i] = vals[i] * sc2;
}

extern "C" void kernel_launch(void* const* d_in, const int* in_sizes, int n_in,
                              void* d_out, int out_size, void* d_ws, size_t ws_size,
                              hipStream_t stream) {
  const float* q     = (const float*)d_in[0];
  const float* k     = (const float*)d_in[1];
  const float* v     = (const float*)d_in[2];
  const float* w_bar = (const float*)d_in[3];
  const float* Ft    = (const float*)d_in[4];
  const float* masks = (const float*)d_in[5];
  float* out = (float*)d_out;
  float* ws  = (float*)d_ws;

  float* kv     = ws + OFF_KV;
  float* Uo     = ws + OFF_U;
  float* Wt     = ws + OFF_WT;
  float* So     = ws + OFF_S;
  float* Ssum   = ws + OFF_SSUM;
  float* gpart  = ws + OFF_GPART;
  float* colsum = ws + OFF_COLSUM;
  float* cc     = ws + OFF_CC;
  float* A0     = ws + OFF_A0;
  float* Lch    = ws + OFF_LCH;
  float* raw    = ws + OFF_RAW;

  hipMemsetAsync(gpart, 0, 2 * 96 * 64 * sizeof(float), stream);
  hipMemsetAsync(raw, 0, BATCH * SEQ * sizeof(float), stream);

  const int NBH = BATCH * HEADS;                 // 96
  kv_kernel<<<NBH, 256, 0, stream>>>(k, v, kv);
  svd_kernel<<<NBH, 256, 0, stream>>>(kv, Uo, Wt, So, Ssum);
  zproj_kernel<<<NBH * 16, 256, 0, stream>>>(q, w_bar, Uo, Wt, gpart, colsum);
  amat_kernel<<<1, 256, 0, stream>>>(masks, A0, Lch);
  chol_kernel<<<1, 64, 0, stream>>>(Lch);
  spec_kernel<<<NBH, 64, 0, stream>>>(gpart, So, Ft, colsum, Ssum, A0, Lch, cc);
  out_kernel<<<NBH * 16, 256, 0, stream>>>(q, Uo, cc, raw);
  norm_kernel<<<BATCH, 256, 0, stream>>>(raw, Ft, out);
}
